// RV2MVImgAttn_82308753260562
// MI455X (gfx1250) — matrix operand order repaired and checked by reading, not compiled, not run on hardware
//
#include <hip/hip_runtime.h>
#include <cstdint>
#include <cstddef>

// ---------------- CDNA5 WMMA / TDM types ----------------
typedef __attribute__((ext_vector_type(16))) __bf16 v16bf;
typedef __attribute__((ext_vector_type(8)))  float  v8f;
typedef __attribute__((ext_vector_type(4)))  unsigned int u32x4;
typedef __attribute__((ext_vector_type(8)))  int    i32x8;
typedef __attribute__((ext_vector_type(4)))  int    i32x4;

#if defined(__HIP_DEVICE_COMPILE__) && __has_builtin(__builtin_amdgcn_tensor_load_to_lds)
#define USE_TDM 1
#else
#define USE_TDM 0
#endif

#define NQ     8192
#define EMBED  128
#define HEADS  8
#define LEVELS 4
#define POINTS 8
#define NCAMS  6
#define NVAL   30825
#define VROWS  (NCAMS * NVAL)

__device__ __forceinline__ uint16_t f2bf(float f) {
  uint32_t u = __float_as_uint(f);
  uint32_t r = u + 0x7FFFu + ((u >> 16) & 1u);   // round-to-nearest-even
  return (uint16_t)(r >> 16);
}

__device__ __forceinline__ float gelu_f(float x) {
  return 0.5f * x * (1.0f + erff(x * 0.70710678118654752440f));
}

// ---------------- weight prep: f32 -> bf16, optional transpose to [N][K] ----------------
__global__ void k_convert_bf16(const float* __restrict__ src, uint16_t* __restrict__ dst,
                               int rows, int cols, int trans) {
  int i = blockIdx.x * 256 + threadIdx.x;
  if (i >= rows * cols) return;
  int r = i / cols, c = i - r * cols;
  uint16_t b = f2bf(src[i]);
  if (trans) dst[(size_t)c * rows + r] = b;   // dst is [cols][rows] = [N][K]
  else       dst[i] = b;                      // already [N][K]
}

// ---------------- generic bf16 WMMA GEMM, 64-row M-tile, TDM-staged A ----------------
// C[M x N] = act(A[M x K] * Wt[N x K]^T + bias) (+ resid), f32 I/O, bf16 matrix core.
// a_trans: A element (m,c) read from A[c*lda + m]. c_trans: store C[n*ldc + m].
// grid = (ceil(M/64), N/128); block = 256 = 8 wave32s; wave w owns col tile w,
// and carries 4 accumulators (rows t*16..t*16+15) so each B fragment feeds 4 WMMAs.
template <int K>
__global__ __launch_bounds__(256)
void k_gemm_bf16(const float* __restrict__ A, int lda, int a_trans, int M,
                 const uint16_t* __restrict__ Wt,
                 const float* __restrict__ bias, int act,
                 const float* __restrict__ resid,
                 float* __restrict__ C, int ldc, int c_trans) {
#if USE_TDM
  __shared__ float    A32[64 * K];            // TDM landing pad (f32 tile)
#endif
  __shared__ uint16_t A_s[64][K + 4];         // bf16 tile, padded vs bank conflicts
  const int m0  = blockIdx.x * 64;
  const int tid = threadIdx.x;

#if USE_TDM
  if ((tid >> 5) == 0) {
    // Build Tensor DMA Descriptor (D#): 2D tile of f32, OOB rows zero-fill the tail.
    uint64_t gaddr = (uint64_t)(uintptr_t)(a_trans ? (A + m0) : (A + (size_t)m0 * lda));
    uint32_t ldsoff = (uint32_t)(uintptr_t)(&A32[0]);
    uint32_t dim0, dim1, tile0, tile1;
    if (a_trans) { dim0 = 64u;        dim1 = (uint32_t)K;        tile0 = 64u;        tile1 = (uint32_t)K; }
    else         { dim0 = (uint32_t)K; dim1 = (uint32_t)(M - m0); tile0 = (uint32_t)K; tile1 = 64u; }
    u32x4 g0;
    g0[0] = 1u;                                             // count=1, user descriptor
    g0[1] = ldsoff;                                         // lds_addr
    g0[2] = (uint32_t)gaddr;                                // global_addr[31:0]
    g0[3] = (uint32_t)((gaddr >> 32) & 0x01FFFFFFu) | (2u << 30);  // addr[56:32] | type=2
    i32x8 g1;
    g1[0] = (int)(2u << 16);                                // data_size = 4 bytes
    g1[1] = (int)((dim0 & 0xFFFFu) << 16);                  // tensor_dim0[15:0]
    g1[2] = (int)((dim0 >> 16) | ((dim1 & 0xFFFFu) << 16)); // dim0[31:16] | dim1[15:0]
    g1[3] = (int)((dim1 >> 16) | (tile0 << 16));            // dim1[31:16] | tile_dim0
    g1[4] = (int)tile1;                                     // tile_dim1 (tile_dim2 = 0)
    g1[5] = (int)lda;                                       // tensor_dim0_stride[31:0]
    g1[6] = 0;
    g1[7] = 0;
    i32x4 z4 = {0, 0, 0, 0};
    i32x8 z8 = {0, 0, 0, 0, 0, 0, 0, 0};
    __builtin_amdgcn_tensor_load_to_lds(g0, g1, z4, z4, z8, 0);
    __builtin_amdgcn_s_wait_tensorcnt(0);
  }
  __syncthreads();
#endif

  // f32 -> bf16 conversion into the WMMA-friendly tile
  for (int idx = tid; idx < 64 * K; idx += 256) {
    int r, c;
    if (a_trans) { r = idx & 63; c = idx >> 6; }
    else         { r = idx / K;  c = idx - r * K; }
#if USE_TDM
    float v = a_trans ? A32[c * 64 + r] : A32[r * K + c];
#else
    int m = m0 + r;
    float v = 0.0f;
    if (m < M) v = a_trans ? A[(size_t)c * lda + m] : A[(size_t)m * lda + c];
#endif
    A_s[r][c] = f2bf(v);
  }
  __syncthreads();

  const int wave = tid >> 5, lane = tid & 31;
  const int half = lane >> 4, lr = lane & 15;
  const int n = blockIdx.y * 128 + wave * 16 + lr;

  v8f acc[4];
  #pragma unroll
  for (int t = 0; t < 4; ++t)
    #pragma unroll
    for (int r = 0; r < 8; ++r) acc[t][r] = 0.0f;

  const uint32_t* Bu = reinterpret_cast<const uint32_t*>(Wt + (size_t)n * K);

  #pragma unroll
  for (int K0 = 0; K0 < K; K0 += 32) {
    union Frag { v16bf v; uint32_t u[8]; };
    // B 32x16 bf16 layout: lanes0-15 K=0-7 & 16-23 ; lanes16-31 K=8-15 & 24-31 (pairs per VGPR)
    Frag b;
    const int kdB = (K0 + (half << 3)) >> 1;
    #pragma unroll
    for (int j = 0; j < 8; ++j) b.u[j] = Bu[kdB + j + (j & 4)];
    // A 16x32 bf16 layout: lanes0-15 K=0-7 & 16-23 ; lanes16-31 K=8-15 & 24-31
    const int kdA = (K0 + (half ? 8 : 0)) >> 1;
    #pragma unroll
    for (int t = 0; t < 4; ++t) {
      Frag a;
      const uint32_t* Au = reinterpret_cast<const uint32_t*>(&A_s[t * 16 + lr][0]);
      #pragma unroll
      for (int i = 0; i < 4; ++i) { a.u[i] = Au[kdA + i]; a.u[4 + i] = Au[kdA + 8 + i]; }
      acc[t] = __builtin_amdgcn_wmma_f32_16x16x32_bf16(false, a.v, false, b.v,
                                                       (short)0, acc[t], false, false);
    }
  }

  const float bn = bias ? bias[n] : 0.0f;
  #pragma unroll
  for (int t = 0; t < 4; ++t) {
    #pragma unroll
    for (int r = 0; r < 8; ++r) {
      int m = m0 + t * 16 + (half ? 8 : 0) + r;   // C: lanes0-15 rows0-7, lanes16-31 rows8-15
      if (m >= M) continue;
      float v = acc[t][r] + bn;
      if (act == 1) v = gelu_f(v);
      else if (act == 2) v = fmaxf(v, 0.0f);
      if (resid) v += resid[(size_t)m * ldc + n];
      if (c_trans) C[(size_t)n * ldc + m] = v;
      else         C[(size_t)m * ldc + n] = v;
    }
  }
}

// ---------------- LayerNorm over 128 cols, one wave per row ----------------
__global__ __launch_bounds__(256)
void k_layernorm128(const float* __restrict__ x, const float* __restrict__ g,
                    const float* __restrict__ b, float* __restrict__ y, int rows) {
  int row = blockIdx.x * 8 + (threadIdx.x >> 5);
  int lane = threadIdx.x & 31;
  if (row >= rows) return;
  float4 v = reinterpret_cast<const float4*>(x + (size_t)row * 128)[lane];
  float s = v.x + v.y + v.z + v.w;
  float q = v.x * v.x + v.y * v.y + v.z * v.z + v.w * v.w;
  for (int d = 16; d; d >>= 1) { s += __shfl_xor(s, d, 32); q += __shfl_xor(q, d, 32); }
  float mean = s * (1.0f / 128.0f);
  float var  = q * (1.0f / 128.0f) - mean * mean;
  float inv  = rsqrtf(var + 1e-5f);
  float4 gg = reinterpret_cast<const float4*>(g)[lane];
  float4 bb = reinterpret_cast<const float4*>(b)[lane];
  float4 o;
  o.x = (v.x - mean) * inv * gg.x + bb.x;
  o.y = (v.y - mean) * inv * gg.y + bb.y;
  o.z = (v.z - mean) * inv * gg.z + bb.z;
  o.w = (v.w - mean) * inv * gg.w + bb.w;
  reinterpret_cast<float4*>(y + (size_t)row * 128)[lane] = o;
}

// ---------------- softmax over 32 (per query, per head), one wave per row ----------------
__global__ __launch_bounds__(256)
void k_softmax32(float* __restrict__ x, int rows) {
  int row = blockIdx.x * 8 + (threadIdx.x >> 5);
  int lane = threadIdx.x & 31;
  if (row >= rows) return;
  float v = x[(size_t)row * 32 + lane];
  float m = v;
  for (int d = 16; d; d >>= 1) m = fmaxf(m, __shfl_xor(m, d, 32));
  float e = expf(v - m);
  float s = e;
  for (int d = 16; d; d >>= 1) s += __shfl_xor(s, d, 32);
  x[(size_t)row * 32 + lane] = e / s;
}

// ---------------- depth head + spherical unprojection + camera projection ----------------
__global__ __launch_bounds__(256)
void k_points(const float* __restrict__ rv, const float* __restrict__ dw,
              const float* __restrict__ db, const float* __restrict__ l2i,
              const int* __restrict__ imgsh, float* __restrict__ rp,
              int* __restrict__ valid, float* __restrict__ depth_out) {
  int n = blockIdx.x * 256 + threadIdx.x;
  if (n >= NQ) return;
  float d = db[0];
  #pragma unroll 8
  for (int c = 0; c < 64; ++c) d += rv[(size_t)c * NQ + n] * dw[c];
  float depth = 55.0f / (1.0f + expf(-d));
  int h = n >> 8, w = n & 255;
  const float PI = 3.14159265358979323846f;
  float az = PI - w * (2.0f * PI / 255.0f);
  float el = 0.17453292519943295f + h * ((-0.5235987755982988f - 0.17453292519943295f) / 31.0f);
  float ce = cosf(el);
  float px = depth * ce * cosf(az);
  float py = depth * ce * sinf(az);
  float pz = depth * sinf(el);
  float Wd = (float)imgsh[1], Hd = (float)imgsh[0];
  for (int cam = 0; cam < NCAMS; ++cam) {
    const float* M = l2i + cam * 16;
    float r0 = M[0] * px + M[1] * py + M[2]  * pz + M[3];
    float r1 = M[4] * px + M[5] * py + M[6]  * pz + M[7];
    float r2 = M[8] * px + M[9] * py + M[10] * pz + M[11];
    bool  mz = r2 > 1e-5f;
    float dz = fmaxf(r2, 1e-5f);
    float rx = (r0 / dz) / Wd;
    float ry = (r1 / dz) / Hd;
    bool ok = mz && (ry > 0.0f) && (ry < 1.0f) && (rx > 0.0f) && (rx < 1.0f);
    rp[((size_t)cam * NQ + n) * 2 + 0] = rx;
    rp[((size_t)cam * NQ + n) * 2 + 1] = ry;
    valid[(size_t)cam * NQ + n] = ok ? 1 : 0;
  }
  depth_out[n] = depth;
}

// ---------------- deformable bilinear sampling; one wave per (cam, query) ----------------
__global__ __launch_bounds__(256)
void k_sample(const float* __restrict__ off, const float* __restrict__ aw,
              const float* __restrict__ rp, const int* __restrict__ valid,
              const float* __restrict__ v, float* __restrict__ samp) {
  int wg = blockIdx.x * 8 + (threadIdx.x >> 5);
  int lane = threadIdx.x & 31;
  int cam = wg >> 13, n = wg & (NQ - 1);
  if (cam >= NCAMS) return;
  if (!valid[(size_t)cam * NQ + n]) return;     // masked out downstream anyway
  int h = lane >> 2, dq = lane & 3;             // 8 heads x 4 lanes, 4 dims/lane
  float rx = rp[((size_t)cam * NQ + n) * 2 + 0];
  float ry = rp[((size_t)cam * NQ + n) * 2 + 1];
  const float* offrow = off + (size_t)n * 512 + h * 64;
  const float* awrow  = aw  + (size_t)n * 256 + h * 32;
  const int LH[4] = {116, 58, 29, 15};
  const int LW[4] = {200, 100, 50, 25};
  const int LS[4] = {0, 23200, 29000, 30450};
  float a0 = 0.f, a1 = 0.f, a2 = 0.f, a3 = 0.f;
  #pragma unroll
  for (int l = 0; l < LEVELS; ++l) {
    const int Hl = LH[l], Wl = LW[l];
    const float* vl = v + ((size_t)cam * NVAL + LS[l]) * 128 + h * 16 + dq * 4;
    #pragma unroll
    for (int p = 0; p < POINTS; ++p) {
      float ox  = offrow[(l * 8 + p) * 2 + 0];
      float oy  = offrow[(l * 8 + p) * 2 + 1];
      float awv = awrow[l * 8 + p];
      float xx = rx * (float)Wl + ox - 0.5f;    // (rx + ox/Wl)*Wl - 0.5
      float yy = ry * (float)Hl + oy - 0.5f;
      float xf = floorf(xx), yf = floorf(yy);
      float fx = xx - xf, fy = yy - yf;
      int xi = (int)xf, yi = (int)yf;
      float s0 = 0.f, s1 = 0.f, s2 = 0.f, s3 = 0.f;
      #pragma unroll
      for (int cy = 0; cy < 2; ++cy) {
        int yc = yi + cy;
        if (yc < 0 || yc >= Hl) continue;
        float wy = cy ? fy : (1.0f - fy);
        #pragma unroll
        for (int cx = 0; cx < 2; ++cx) {
          int xc = xi + cx;
          if (xc < 0 || xc >= Wl) continue;
          float wgt = wy * (cx ? fx : (1.0f - fx));
          const float4 gv = *reinterpret_cast<const float4*>(vl + (size_t)(yc * Wl + xc) * 128);
          s0 += wgt * gv.x; s1 += wgt * gv.y; s2 += wgt * gv.z; s3 += wgt * gv.w;
        }
      }
      a0 += s0 * awv; a1 += s1 * awv; a2 += s2 * awv; a3 += s3 * awv;
    }
  }
  float* o = samp + ((size_t)cam * NQ + n) * 128 + h * 16 + dq * 4;
  o[0] = a0; o[1] = a1; o[2] = a2; o[3] = a3;
}

// ---------------- masked camera combine: slots = sum_c valid*samp / max(count,1) ----------------
__global__ __launch_bounds__(256)
void k_combine(const float* __restrict__ samp, const int* __restrict__ valid,
               float* __restrict__ slots) {
  int gid = blockIdx.x * 256 + threadIdx.x;
  if (gid >= NQ * 128) return;
  int n = gid >> 7, k = gid & 127;
  float s = 0.0f; int cnt = 0;
  #pragma unroll
  for (int c = 0; c < NCAMS; ++c) {
    if (valid[(size_t)c * NQ + n]) { s += samp[((size_t)c * NQ + n) * 128 + k]; ++cnt; }
  }
  slots[gid] = s / (float)(cnt > 0 ? cnt : 1);
}

// =====================================================================================
extern "C" void kernel_launch(void* const* d_in, const int* in_sizes, int n_in,
                              void* d_out, int out_size, void* d_ws, size_t ws_size,
                              hipStream_t stream) {
  (void)in_sizes; (void)n_in; (void)out_size; (void)ws_size;
  const float* in_rv  = (const float*)d_in[0];
  const float* mlvl   = (const float*)d_in[1];
  const float* l2i    = (const float*)d_in[4];
  const int*   imgsh  = (const int*)d_in[5];
  const float* q_w1 = (const float*)d_in[6],  *q_b1 = (const float*)d_in[7];
  const float* q_w2 = (const float*)d_in[8],  *q_b2 = (const float*)d_in[9];
  const float* dw   = (const float*)d_in[10], *db   = (const float*)d_in[11];
  const float* ln1g = (const float*)d_in[12], *ln1b = (const float*)d_in[13];
  const float* ln2g = (const float*)d_in[14], *ln2b = (const float*)d_in[15];
  const float* Wv   = (const float*)d_in[16], *bv   = (const float*)d_in[17];
  const float* Woff = (const float*)d_in[18], *boff = (const float*)d_in[19];
  const float* Waw  = (const float*)d_in[20], *baw  = (const float*)d_in[21];
  const float* Wout = (const float*)d_in[22], *bout = (const float*)d_in[23];
  const float* f1w  = (const float*)d_in[24], *f1b  = (const float*)d_in[25];
  const float* f2w  = (const float*)d_in[26], *f2b  = (const float*)d_in[27];
  const float* o_w1 = (const float*)d_in[28], *o_b1 = (const float*)d_in[29];
  const float* o_w2 = (const float*)d_in[30], *o_b2 = (const float*)d_in[31];

  char* base = (char*)d_ws;
  size_t off = 0;
  auto alloc = [&](size_t bytes) -> void* {
    void* p = base + off;
    off = (off + bytes + 255) & ~(size_t)255;
    return p;
  };
  uint16_t* wt_q1  = (uint16_t*)alloc(128 * 64 * 2);
  uint16_t* wt_q2  = (uint16_t*)alloc(128 * 128 * 2);
  uint16_t* wt_v   = (uint16_t*)alloc(128 * 128 * 2);
  uint16_t* wt_off = (uint16_t*)alloc(512 * 128 * 2);
  uint16_t* wt_aw  = (uint16_t*)alloc(256 * 128 * 2);
  uint16_t* wt_out = (uint16_t*)alloc(128 * 128 * 2);
  uint16_t* wt_f1  = (uint16_t*)alloc(128 * 128 * 2);
  uint16_t* wt_f2  = (uint16_t*)alloc(128 * 128 * 2);
  uint16_t* wt_o1  = (uint16_t*)alloc(128 * 128 * 2);
  uint16_t* wt_o2  = (uint16_t*)alloc(128 * 128 * 2);
  float* rv1   = (float*)alloc((size_t)NQ * 128 * 4);      // reused as FFN hidden
  float* q_ws  = (float*)alloc((size_t)NQ * 128 * 4);      // reused as fused
  float* qn    = (float*)alloc((size_t)NQ * 128 * 4);
  float* offw  = (float*)alloc((size_t)NQ * 512 * 4);      // reused as t1
  float* aww   = (float*)alloc((size_t)NQ * 256 * 4);      // reused as slots
  float* rpw   = (float*)alloc((size_t)NCAMS * NQ * 2 * 4);
  int*   vld   = (int*)  alloc((size_t)NCAMS * NQ * 4);
  float* v_ws  = (float*)alloc((size_t)VROWS * 128 * 4);
  float* samp  = (float*)alloc((size_t)NCAMS * NQ * 128 * 4);
  float* nf    = (float*)alloc((size_t)NQ * 128 * 4);
  float* y_ws  = (float*)alloc((size_t)NQ * 128 * 4);
  float* slots = aww;            // aw consumed by sampling before combine writes
  float* fused = q_ws;           // pre-LN q dead after LN1
  float* hid   = rv1;            // rv1 dead after GEMM2
  float* t1    = offw;           // off consumed by sampling before out-convs

  float* out   = (float*)d_out;                 // [128][8192]
  float* depth = (float*)d_out + (size_t)128 * NQ;

  auto conv = [&](const float* s, uint16_t* d, int r, int c, int t) {
    int nthr = r * c;
    k_convert_bf16<<<(nthr + 255) / 256, 256, 0, stream>>>(s, d, r, c, t);
  };
  // [O][C] weights keep layout; [K][N] weights get transposed to [N][K]
  conv(q_w1, wt_q1, 128, 64, 0);
  conv(q_w2, wt_q2, 128, 128, 0);
  conv(Wv,   wt_v,  128, 128, 1);
  conv(Woff, wt_off, 128, 512, 1);
  conv(Waw,  wt_aw,  128, 256, 1);
  conv(Wout, wt_out, 128, 128, 1);
  conv(f1w,  wt_f1,  128, 128, 1);
  conv(f2w,  wt_f2,  128, 128, 1);
  conv(o_w1, wt_o1,  128, 128, 0);
  conv(o_w2, wt_o2,  128, 128, 0);

  // depth head + projection (also writes depths output)
  k_points<<<NQ / 256, 256, 0, stream>>>(in_rv, dw, db, l2i, imgsh, rpw, vld, depth);

  // query MLP: gelu(X^T W1^T + b1) W2^T + b2
  k_gemm_bf16<64><<<dim3(NQ / 64, 1), 256, 0, stream>>>(in_rv, NQ, 1, NQ, wt_q1, q_b1, 1, nullptr, rv1, 128, 0);
  k_gemm_bf16<128><<<dim3(NQ / 64, 1), 256, 0, stream>>>(rv1, 128, 0, NQ, wt_q2, q_b2, 0, nullptr, q_ws, 128, 0);
  k_layernorm128<<<NQ / 8, 256, 0, stream>>>(q_ws, ln1g, ln1b, qn, NQ);

  // value projection over all cameras (bulk of GEMM FLOPs; output L2-resident for gather)
  k_gemm_bf16<128><<<dim3((VROWS + 63) / 64, 1), 256, 0, stream>>>(mlvl, 128, 0, VROWS, wt_v, bv, 0, nullptr, v_ws, 128, 0);

  // sampling offsets + attention weights (shared across cameras: q_rb is qn broadcast)
  k_gemm_bf16<128><<<dim3(NQ / 64, 4), 256, 0, stream>>>(qn, 128, 0, NQ, wt_off, boff, 0, nullptr, offw, 512, 0);
  k_gemm_bf16<128><<<dim3(NQ / 64, 2), 256, 0, stream>>>(qn, 128, 0, NQ, wt_aw, baw, 0, nullptr, aww, 256, 0);
  k_softmax32<<<(NQ * HEADS) / 8, 256, 0, stream>>>(aww, NQ * HEADS);

  // deformable gather + masked camera reduce
  k_sample<<<(NCAMS * NQ) / 8, 256, 0, stream>>>(offw, aww, rpw, vld, v_ws, samp);
  k_combine<<<(NQ * 128) / 256, 256, 0, stream>>>(samp, vld, slots);

  // output proj + residual(qn), LN2, FFN, residual(nf)
  k_gemm_bf16<128><<<dim3(NQ / 64, 1), 256, 0, stream>>>(slots, 128, 0, NQ, wt_out, bout, 0, qn, fused, 128, 0);
  k_layernorm128<<<NQ / 8, 256, 0, stream>>>(fused, ln2g, ln2b, nf, NQ);
  k_gemm_bf16<128><<<dim3(NQ / 64, 1), 256, 0, stream>>>(nf, 128, 0, NQ, wt_f1, f1b, 2, nullptr, hid, 128, 0);
  k_gemm_bf16<128><<<dim3(NQ / 64, 1), 256, 0, stream>>>(hid, 128, 0, NQ, wt_f2, f2b, 0, nf, y_ws, 128, 0);

  // faithful torch .view(B,C,H,W) reinterpretation -> transposed-A convs, transposed-C store
  k_gemm_bf16<128><<<dim3(NQ / 64, 1), 256, 0, stream>>>(y_ws, NQ, 1, NQ, wt_o1, o_b1, 1, nullptr, t1, 128, 0);
  k_gemm_bf16<128><<<dim3(NQ / 64, 1), 256, 0, stream>>>(t1, 128, 0, NQ, wt_o2, o_b2, 0, nullptr, out, NQ, 1);
}
